// MultiHeadAttention_31877247271022
// MI455X (gfx1250) — compile-verified, matrix operand order checked by reading, not run on hardware
//
#include <hip/hip_runtime.h>

// ---------------------------------------------------------------------------
// Multi-head attention forward for MI455X (gfx1250, wave32, WMMA bf16).
// Pipeline: 3x input-projection GEMM (f32 -> bf16 head-split), flash-style
// causal attention (bf16 WMMA, f32 online softmax), output-projection GEMM.
// Round 3: templated output layout (no runtime branch) + strength-reduced
// epilogue addressing (one base per C tile, constant stride per r).
// ---------------------------------------------------------------------------

typedef __attribute__((ext_vector_type(16))) __bf16 v16bf;
typedef __attribute__((ext_vector_type(8)))  float  v8f;

#define D_MODEL 1024
#define NHEADS  16
#define DEPTH   64
#define BATCH   4
#define SEQ     2048
#define MROWS   (BATCH * SEQ)   // 8192

// Pack two f32 into one dword of bf16 (hardware convert).
__device__ __forceinline__ unsigned pack_bf16x2(float lo, float hi) {
    union { __bf16 h[2]; unsigned u; } p;
    p.h[0] = (__bf16)lo;
    p.h[1] = (__bf16)hi;
    return p.u;
}

__device__ __forceinline__ unsigned short bf16_bits(float f) {
    union { __bf16 h; unsigned short s; } c;
    c.h = (__bf16)f;
    return c.s;
}

// Load one 16x32 bf16 WMMA fragment (A- or B-operand). Caller passes a pointer
// to the 32 contiguous k-elements of this lane's row/column. Fragment element
// e maps to k = 16*(e/8) + 8*half + (e%8)  (half = lane>>4), so consecutive
// element pairs are consecutive k -> 8 dword loads (vectorizes to 2x b128).
template <typename PtrT>
__device__ __forceinline__ v16bf load_frag32(PtrT p, int half) {
    union { v16bf v; unsigned u[8]; } f;
#pragma unroll
    for (int i = 0; i < 8; ++i) {
        int k = ((i >> 2) << 4) + (half << 3) + ((i & 3) << 1);
        f.u[i] = *(const unsigned*)(p + k);
    }
    return f.v;
}

// ---------------------------------------------------------------------------
// Input projection: out = split_heads(X @ W + bias), stored bf16.
// Block tile 64(M) x 128(N), 8 waves (2x4), K-chunks of 32.
// TRANSPOSE_V == 0 : out layout [B, H, S, DEPTH]   (for Q, K)
// TRANSPOSE_V == 1 : out layout [B, H, DEPTH, S]   (for V; PV B-fragment order)
// ---------------------------------------------------------------------------
template <int TRANSPOSE_V>
__global__ __launch_bounds__(256) void proj_bf16_kernel(
    const float* __restrict__ X,      // [MROWS, D_MODEL] f32
    const float* __restrict__ W,      // [D_MODEL, D_MODEL] f32
    const float* __restrict__ bias,   // [D_MODEL] f32
    unsigned short* __restrict__ out) // bf16
{
    __shared__ unsigned AsU[64 * 16];     // [m][k/2]   bf16 pairs
    __shared__ unsigned BsU[128 * 16];    // [n][k/2]   (transposed for frag load)

    const int tid   = threadIdx.x;
    const int wave  = tid >> 5;
    const int lane  = tid & 31;
    const int half  = lane >> 4;
    const int ln    = lane & 15;
    const int waveM = wave >> 2;   // 0..1
    const int waveN = wave & 3;    // 0..3
    const int m0    = blockIdx.y * 64;
    const int n0    = blockIdx.x * 128;

    // staging coordinates (A tile: 4 threads/row, 8 f32 each)
    const int ar = tid >> 2;            // 0..63
    const int ac = (tid & 3) << 3;      // 0,8,16,24

    v8f acc[2][2];
    v8f vz = {};
#pragma unroll
    for (int a = 0; a < 2; ++a)
#pragma unroll
        for (int b = 0; b < 2; ++b) acc[a][b] = vz;

    for (int kt = 0; kt < D_MODEL; kt += 32) {
        __syncthreads();
        // ---- A tile: two b128 loads + one b128 LDS store per thread ----
        {
            const float* xp = &X[(size_t)(m0 + ar) * D_MODEL + kt + ac];
            union { float4 v; float f[4]; } a0, a1;
            a0.v = *(const float4*)xp;
            a1.v = *(const float4*)(xp + 4);
            uint4 pk;
            pk.x = pack_bf16x2(a0.f[0], a0.f[1]);
            pk.y = pack_bf16x2(a0.f[2], a0.f[3]);
            pk.z = pack_bf16x2(a1.f[0], a1.f[1]);
            pk.w = pack_bf16x2(a1.f[2], a1.f[3]);
            *(uint4*)&AsU[ar * 16 + (ac >> 1)] = pk;
        }
        // ---- B tile: 4 n-cols x 2 k-rows per unit; pack k-pairs into dwords ----
#pragma unroll
        for (int jj = 0; jj < 2; ++jj) {
            int u  = tid + 256 * jj;          // 0..511
            int n  = (u & 31) << 2;           // 0,4,...,124
            int kp = u >> 5;                  // 0..15  (k pair index)
            const float* wp = &W[(size_t)(kt + (kp << 1)) * D_MODEL + n0 + n];
            union { float4 v; float f[4]; } r0, r1;
            r0.v = *(const float4*)wp;
            r1.v = *(const float4*)(wp + D_MODEL);
#pragma unroll
            for (int qq = 0; qq < 4; ++qq)
                BsU[(n + qq) * 16 + kp] = pack_bf16x2(r0.f[qq], r1.f[qq]);
        }
        __syncthreads();

        v16bf af[2], bfr[2];
#pragma unroll
        for (int t = 0; t < 2; ++t)
            af[t] = load_frag32((const unsigned short*)AsU + (waveM * 32 + t * 16 + ln) * 32, half);
#pragma unroll
        for (int t = 0; t < 2; ++t)
            bfr[t] = load_frag32((const unsigned short*)BsU + (waveN * 32 + t * 16 + ln) * 32, half);
#pragma unroll
        for (int a = 0; a < 2; ++a)
#pragma unroll
            for (int b = 0; b < 2; ++b)
                acc[a][b] = __builtin_amdgcn_wmma_f32_16x16x32_bf16(
                    false, af[a], false, bfr[b], (short)0, acc[a][b], false, false);
    }

    // Epilogue: one base index per 16x16 C tile; r advances by a fixed stride.
    // 64-row blocks never cross a batch boundary (SEQ % 64 == 0).
#pragma unroll
    for (int a = 0; a < 2; ++a) {
        const int mb = m0 + waveM * 32 + a * 16 + 8 * half;  // row at r=0
        const int bb = mb >> 11;          // mb / SEQ
        const int s  = mb & (SEQ - 1);    // mb % SEQ
#pragma unroll
        for (int b = 0; b < 2; ++b) {
            const int n = n0 + waveN * 32 + b * 16 + ln;
            const int h = n >> 6, d = n & (DEPTH - 1);
            const float bi = bias[n];
            size_t base, stride;
            if (TRANSPOSE_V) {
                base   = (((size_t)bb * NHEADS + h) * DEPTH + d) * SEQ + s;
                stride = 1;
            } else {
                base   = (((size_t)bb * NHEADS + h) * SEQ + s) * DEPTH + d;
                stride = DEPTH;
            }
#pragma unroll
            for (int r = 0; r < 8; ++r)
                out[base + (size_t)r * stride] = bf16_bits(acc[a][b][r] + bi);
        }
    }
}

// ---------------------------------------------------------------------------
// Flash-style causal attention. One workgroup per (query-block-of-64, b*h).
// 4 waves, each owning 16 query rows. Logits/softmax/PV never touch HBM.
// Qh, Kh: bf16 [B,H,S,DEPTH];  Vh: bf16 [B,H,DEPTH,S];  O: bf16 [B,S,D_MODEL]
// ---------------------------------------------------------------------------
__global__ __launch_bounds__(128) void attn_kernel(
    const unsigned short* __restrict__ Qh,
    const unsigned short* __restrict__ Kh,
    const unsigned short* __restrict__ Vh,
    unsigned short* __restrict__ O)
{
    __shared__ unsigned short Ps[4][16 * 32];   // per-wave P staging (C-frag -> A-frag)

    const int qb   = blockIdx.x;           // 0..SEQ/64-1
    const int bh   = blockIdx.y;           // 0..B*H-1
    const int b    = bh / NHEADS;
    const int h    = bh % NHEADS;
    const int wave = threadIdx.x >> 5;
    const int lane = threadIdx.x & 31;
    const int half = lane >> 4;
    const int ln   = lane & 15;
    const int q0   = qb * 64 + wave * 16;  // first query row of this wave

    const unsigned short* Qp = Qh + (size_t)bh * SEQ * DEPTH;
    const unsigned short* Kp = Kh + (size_t)bh * SEQ * DEPTH;
    const unsigned short* Vp = Vh + (size_t)bh * DEPTH * SEQ;
    unsigned short* myP = &Ps[wave][0];

    // Q fragments (depth 64 -> two K=32 fragments), loaded once.
    v16bf qfrag[2];
#pragma unroll
    for (int kk = 0; kk < 2; ++kk)
        qfrag[kk] = load_frag32(Qp + (size_t)(q0 + ln) * DEPTH + kk * 32, half);

    v8f o[4];
    v8f vz = {};
#pragma unroll
    for (int j = 0; j < 4; ++j) o[j] = vz;
    float mrow[8], lrow[8];
#pragma unroll
    for (int r = 0; r < 8; ++r) { mrow[r] = -__builtin_inff(); lrow[r] = 0.0f; }

    const float scale = 0.125f;            // 1/sqrt(DEPTH)
    const int qmax = q0 + 15;

    for (int kc = 0; kc <= qmax; kc += 32) {       // causal: skip key chunks > qmax
        // ---- logits: 16 queries x 32 keys (two 16-key tiles) ----
        v8f st[2];
#pragma unroll
        for (int t = 0; t < 2; ++t) {
            const unsigned short* kp = Kp + (size_t)(kc + t * 16 + ln) * DEPTH;
            v16bf kf0 = load_frag32(kp, half);
            v16bf kf1 = load_frag32(kp + 32, half);
            v8f s = vz;
            s = __builtin_amdgcn_wmma_f32_16x16x32_bf16(false, qfrag[0], false, kf0,
                                                        (short)0, s, false, false);
            s = __builtin_amdgcn_wmma_f32_16x16x32_bf16(false, qfrag[1], false, kf1,
                                                        (short)0, s, false, false);
            st[t] = s;
        }
        // scale + causal mask (kj > qi disallowed)
#pragma unroll
        for (int t = 0; t < 2; ++t)
#pragma unroll
        for (int r = 0; r < 8; ++r) {
            int qi = q0 + r + 8 * half;
            int kj = kc + t * 16 + ln;
            float v = st[t][r] * scale;
            st[t][r] = (kj > qi) ? -1.0e9f : v;
        }
        // ---- online softmax (per row r+8*half, reduced over 16 column lanes) ----
        float alpha[8];
#pragma unroll
        for (int r = 0; r < 8; ++r) {
            float mx = fmaxf(st[0][r], st[1][r]);
            mx = fmaxf(mx, __shfl_xor(mx, 1, 32));
            mx = fmaxf(mx, __shfl_xor(mx, 2, 32));
            mx = fmaxf(mx, __shfl_xor(mx, 4, 32));
            mx = fmaxf(mx, __shfl_xor(mx, 8, 32));
            float nm = fmaxf(mrow[r], mx);
            alpha[r] = __expf(mrow[r] - nm);
            mrow[r]  = nm;
            float p0 = __expf(st[0][r] - nm);
            float p1 = __expf(st[1][r] - nm);
            st[0][r] = p0; st[1][r] = p1;
            float sm = p0 + p1;
            sm += __shfl_xor(sm, 1, 32);
            sm += __shfl_xor(sm, 2, 32);
            sm += __shfl_xor(sm, 4, 32);
            sm += __shfl_xor(sm, 8, 32);
            lrow[r] = lrow[r] * alpha[r] + sm;
        }
        // rescale O accumulators
#pragma unroll
        for (int j = 0; j < 4; ++j)
#pragma unroll
            for (int r = 0; r < 8; ++r) o[j][r] *= alpha[r];
        // ---- stage P (bf16) through LDS to get A-fragment layout ----
#pragma unroll
        for (int t = 0; t < 2; ++t)
#pragma unroll
        for (int r = 0; r < 8; ++r)
            myP[(r + 8 * half) * 32 + t * 16 + ln] = bf16_bits(st[t][r]);
        asm volatile("s_wait_dscnt 0x0" ::: "memory");   // per-wave LDS RAW fence
        v16bf pfrag = load_frag32(myP + ln * 32, half);
        // ---- PV: P[16x32] x V[32 x 64] ----
#pragma unroll
        for (int j = 0; j < 4; ++j) {
            v16bf vf = load_frag32(Vp + (size_t)(j * 16 + ln) * SEQ + kc, half);
            o[j] = __builtin_amdgcn_wmma_f32_16x16x32_bf16(false, pfrag, false, vf,
                                                           (short)0, o[j], false, false);
        }
    }

    // ---- normalize and store O as bf16 [B, S, D_MODEL] (merge heads) ----
    float linv[8];
#pragma unroll
    for (int r = 0; r < 8; ++r) linv[r] = 1.0f / lrow[r];
    const size_t base0 = ((size_t)b * SEQ + q0 + 8 * half) * D_MODEL + h * DEPTH + ln;
#pragma unroll
    for (int j = 0; j < 4; ++j)
#pragma unroll
    for (int r = 0; r < 8; ++r)
        O[base0 + (size_t)r * D_MODEL + j * 16] = bf16_bits(o[j][r] * linv[r]);
}

// ---------------------------------------------------------------------------
// Output projection: d_out = O @ Wo + bo  (A already bf16; f32 output).
// ---------------------------------------------------------------------------
__global__ __launch_bounds__(256) void out_proj_kernel(
    const unsigned short* __restrict__ A, // [MROWS, D_MODEL] bf16
    const float* __restrict__ W,          // [D_MODEL, D_MODEL] f32
    const float* __restrict__ bias,       // [D_MODEL] f32
    float* __restrict__ out)              // [MROWS, D_MODEL] f32
{
    __shared__ unsigned AsU[64 * 16];
    __shared__ unsigned BsU[128 * 16];

    const int tid   = threadIdx.x;
    const int wave  = tid >> 5;
    const int lane  = tid & 31;
    const int half  = lane >> 4;
    const int ln    = lane & 15;
    const int waveM = wave >> 2;
    const int waveN = wave & 3;
    const int m0    = blockIdx.y * 64;
    const int n0    = blockIdx.x * 128;

    const int ar  = tid >> 2;           // 0..63
    const int ac4 = (tid & 3) << 2;     // dword col 0,4,8,12

    v8f acc[2][2];
    v8f vz = {};
#pragma unroll
    for (int a = 0; a < 2; ++a)
#pragma unroll
        for (int b = 0; b < 2; ++b) acc[a][b] = vz;

    for (int kt = 0; kt < D_MODEL; kt += 32) {
        __syncthreads();
        // A tile already bf16: one b128 copy per thread
        *(uint4*)&AsU[ar * 16 + ac4] =
            *(const uint4*)&A[(size_t)(m0 + ar) * D_MODEL + kt + (ac4 << 1)];
        // B tile: convert f32 -> bf16 pairs (same scheme as proj kernel)
#pragma unroll
        for (int jj = 0; jj < 2; ++jj) {
            int u  = tid + 256 * jj;
            int n  = (u & 31) << 2;
            int kp = u >> 5;
            const float* wp = &W[(size_t)(kt + (kp << 1)) * D_MODEL + n0 + n];
            union { float4 v; float f[4]; } r0, r1;
            r0.v = *(const float4*)wp;
            r1.v = *(const float4*)(wp + D_MODEL);
#pragma unroll
            for (int qq = 0; qq < 4; ++qq)
                BsU[(n + qq) * 16 + kp] = pack_bf16x2(r0.f[qq], r1.f[qq]);
        }
        __syncthreads();

        v16bf af[2], bfr[2];
#pragma unroll
        for (int t = 0; t < 2; ++t)
            af[t] = load_frag32((const unsigned short*)AsU + (waveM * 32 + t * 16 + ln) * 32, half);
#pragma unroll
        for (int t = 0; t < 2; ++t)
            bfr[t] = load_frag32((const unsigned short*)BsU + (waveN * 32 + t * 16 + ln) * 32, half);
#pragma unroll
        for (int a = 0; a < 2; ++a)
#pragma unroll
            for (int b = 0; b < 2; ++b)
                acc[a][b] = __builtin_amdgcn_wmma_f32_16x16x32_bf16(
                    false, af[a], false, bfr[b], (short)0, acc[a][b], false, false);
    }

#pragma unroll
    for (int a = 0; a < 2; ++a) {
        const int mb = m0 + waveM * 32 + a * 16 + 8 * half;
#pragma unroll
        for (int b = 0; b < 2; ++b) {
            const int n = n0 + waveN * 32 + b * 16 + ln;
            const float bi = bias[n];
            const size_t base = (size_t)mb * D_MODEL + n;
#pragma unroll
            for (int r = 0; r < 8; ++r)
                out[base + (size_t)r * D_MODEL] = acc[a][b][r] + bi;
        }
    }
}

// ---------------------------------------------------------------------------
// Launch. Workspace layout (bf16 elements): Qh | Kh | Vh | O  (4 x 16 MB).
// ---------------------------------------------------------------------------
extern "C" void kernel_launch(void* const* d_in, const int* in_sizes, int n_in,
                              void* d_out, int out_size, void* d_ws, size_t ws_size,
                              hipStream_t stream) {
    (void)in_sizes; (void)n_in; (void)out_size; (void)ws_size;

    const float* q  = (const float*)d_in[0];
    const float* k  = (const float*)d_in[1];
    const float* v  = (const float*)d_in[2];
    /* d_in[3] = causal mask; applied analytically inside attn_kernel */
    const float* wq = (const float*)d_in[4];
    const float* bq = (const float*)d_in[5];
    const float* wk = (const float*)d_in[6];
    const float* bk = (const float*)d_in[7];
    const float* wv = (const float*)d_in[8];
    const float* bv = (const float*)d_in[9];
    const float* wo = (const float*)d_in[10];
    const float* bo = (const float*)d_in[11];

    const size_t HE = (size_t)BATCH * NHEADS * SEQ * DEPTH;   // 8,388,608 elems
    unsigned short* Qh   = (unsigned short*)d_ws;
    unsigned short* Kh   = Qh + HE;
    unsigned short* Vh   = Kh + HE;
    unsigned short* Obuf = Vh + HE;

    dim3 gemmGrid(D_MODEL / 128, MROWS / 64);   // (8, 128)
    proj_bf16_kernel<0><<<gemmGrid, 256, 0, stream>>>(q, wq, bq, Qh);
    proj_bf16_kernel<0><<<gemmGrid, 256, 0, stream>>>(k, wk, bk, Kh);
    proj_bf16_kernel<1><<<gemmGrid, 256, 0, stream>>>(v, wv, bv, Vh);

    attn_kernel<<<dim3(SEQ / 64, BATCH * NHEADS), 128, 0, stream>>>(Qh, Kh, Vh, Obuf);

    out_proj_kernel<<<gemmGrid, 256, 0, stream>>>(Obuf, wo, bo, (float*)d_out);
}
